// SegmentationPostProcess_2113123909851
// MI455X (gfx1250) — matrix operand order repaired and checked by reading, not compiled, not run on hardware
//
#include <hip/hip_runtime.h>
#include <hip/hip_bf16.h>

// Problem constants (from the reference setup)
#define B_   4
#define Q_   100
#define C_   80          // 5 tiles of 16
#define HM_  128
#define WM_  128
#define HW_  (HM_ * WM_) // 16384 pixels per (b,q) mask
#define OUT_ 512
#define KTILES_ 4        // K = 128 = 4 * 32 (Q=100 zero-padded)
#define MTILES_ 5        // C = 80  = 5 * 16

typedef __attribute__((ext_vector_type(16))) _Float16 v16h;
typedef __attribute__((ext_vector_type(8)))  float    v8f;

// Workspace layout: M[B][C][128][128] f32, then A-fragments (f16, fragment order)
#define M_BYTES ((size_t)B_ * C_ * HW_ * sizeof(float))          // ~21 MB, L2-resident
#define AFRAG_HALVES_PER_B (KTILES_ * MTILES_ * 32 * 16)         // 10240

// Fast sigmoid: v_mul + v_exp_f32 + v_add + v_rcp_f32 (1-ULP rcp instead of the
// precise div sequence; result is f16-quantized and only feeds an argmax).
__device__ __forceinline__ float fast_sigmoid(float x) {
    return __builtin_amdgcn_rcpf(1.0f + __expf(-x));
}

// ---------------------------------------------------------------------------
// Kernel 1: per-query class argmax, class counts -> 1/max(cnt,1), and bake the
// scaled one-hot A matrix directly in v_wmma_f32_16x16x32_f16 A-fragment order:
//   lane L: row m = L&15, half h = L>>4; element e<8 -> K = 8h+e,
//   e>=8 -> K = 16 + 8h + (e-8). (ISA 7.12.2, 16-bit A 16x32 layout)
// ---------------------------------------------------------------------------
__global__ void segpost_build_assign(const float* __restrict__ logits,
                                     _Float16* __restrict__ afrag) {
    const int b = blockIdx.x;
    const int t = threadIdx.x;
    __shared__ int   cls[Q_];
    __shared__ int   cnt[C_];
    __shared__ float inv[C_];

    if (t < C_) cnt[t] = 0;
    __syncthreads();

    if (t < Q_) {
        const float* lp = logits + ((size_t)b * Q_ + t) * C_;
        float best = lp[0];
        int   bi   = 0;
        #pragma unroll 4
        for (int c = 1; c < C_; ++c) {
            float v = lp[c];
            if (v > best) { best = v; bi = c; }   // first max wins (strict >)
        }
        cls[t] = bi;
        atomicAdd(&cnt[bi], 1);
    }
    __syncthreads();

    if (t < C_) inv[t] = __builtin_amdgcn_rcpf(fmaxf((float)cnt[t], 1.0f));
    __syncthreads();

    // i = ((kt*5 + mt)*32 + lane)*16 + e
    for (int i = t; i < AFRAG_HALVES_PER_B; i += blockDim.x) {
        int e    = i & 15;
        int lane = (i >> 4) & 31;
        int rest = i >> 9;           // kt*5 + mt
        int mt   = rest % MTILES_;
        int kt   = rest / MTILES_;
        int m    = lane & 15;
        int h    = lane >> 4;
        int c    = mt * 16 + m;
        int kk   = (e < 8) ? (h * 8 + e) : (16 + h * 8 + (e - 8));
        int q    = kt * 32 + kk;
        _Float16 v = (_Float16)0.0f;
        if (q < Q_ && cls[q] == c) v = (_Float16)inv[c];
        afrag[(((size_t)b * KTILES_ + kt) * MTILES_ + mt) * 32 * 16
              + (size_t)lane * 16 + e] = v;
    }
}

// ---------------------------------------------------------------------------
// Kernel 2: segment-mean as WMMA GEMM.
//   M[b, 0:80, pixTile] = A[80x128] x sigmoid(masks)[128 x 16]
// One wave per 16-pixel N-tile; B fragments loaded once (sigmoid on the fly,
// f16), reused across 5 class tiles -> 20 v_wmma_f32_16x16x32_f16 per wave.
// B 32x16 layout: n = lane&15, K = 8*(lane>>4) + e for e<8, 8 + 8*(lane>>4) + e for e>=8. Guards are
// branch-free selects so EXEC stays all-ones at every WMMA.
// ---------------------------------------------------------------------------
__global__ void __launch_bounds__(256)
segpost_wmma_segmean(const float* __restrict__ masks,
                     const _Float16* __restrict__ afrag,
                     float* __restrict__ Mbuf) {
    const int wave = blockIdx.x * (blockDim.x >> 5) + (threadIdx.x >> 5);
    const int lane = threadIdx.x & 31;
    const int b    = wave >> 10;        // 1024 N-tiles per batch
    const int tile = wave & 1023;
    const int n    = lane & 15;
    const int kh   = (lane >> 4) & 1;
    const int pix  = (tile << 4) + n;

    // Build B fragments: 4 k-steps of 32 queries (zero-padded past Q=100)
    v16h bfrag[KTILES_];
    #pragma unroll
    for (int kt = 0; kt < KTILES_; ++kt) {
        #pragma unroll
        for (int e = 0; e < 16; ++e) {
            int   q   = kt * 32 + ((e < 8) ? (kh * 8 + e) : (16 + kh * 8 + (e - 8)));
            int   qc  = (q < Q_) ? q : (Q_ - 1);          // clamp: always-valid load
            float mv  = masks[((size_t)b * Q_ + qc) * HW_ + pix];
            float sg  = fast_sigmoid(mv);                 // mul+exp+add+rcp
            bfrag[kt][e] = (q < Q_) ? (_Float16)sg : (_Float16)0.0f;
        }
    }

    v8f acc[MTILES_];
    #pragma unroll
    for (int mt = 0; mt < MTILES_; ++mt) acc[mt] = (v8f){0,0,0,0,0,0,0,0};

    const _Float16* ab = afrag + (size_t)b * AFRAG_HALVES_PER_B + (size_t)lane * 16;
    #pragma unroll
    for (int mt = 0; mt < MTILES_; ++mt) {
        #pragma unroll
        for (int kt = 0; kt < KTILES_; ++kt) {
            v16h a = *(const v16h*)(ab + ((size_t)kt * MTILES_ + mt) * 32 * 16);
            acc[mt] = __builtin_amdgcn_wmma_f32_16x16x32_f16(
                /*neg_a=*/false, a, /*neg_b=*/false, bfrag[kt],
                /*c_mod=*/(short)0, acc[mt], /*reuse_a=*/false, /*reuse_b=*/false);
        }
    }

    // C/D layout: VGPR r -> class row r + 8*kh within the tile, column n.
    #pragma unroll
    for (int mt = 0; mt < MTILES_; ++mt) {
        #pragma unroll
        for (int r = 0; r < 8; ++r) {
            int c = mt * 16 + kh * 8 + r;
            Mbuf[((size_t)b * C_ + c) * HW_ + pix] = acc[mt][r];
        }
    }
}

// ---------------------------------------------------------------------------
// Kernel 3: fused bilinear x4 upsample (half-pixel, edge clamp; the second
// 512->512 resize in the reference is identity) + argmax over 80 classes.
// M is L2-resident (21 MB << 192 MB), so the 80x4 corner reads hit L2.
// ---------------------------------------------------------------------------
__global__ void __launch_bounds__(256)
segpost_upsample_argmax(const float* __restrict__ Mbuf,
                        int* __restrict__ out) {
    const int idx = blockIdx.x * blockDim.x + threadIdx.x;   // B*512*512 exact
    const int x = idx & (OUT_ - 1);
    const int y = (idx >> 9) & (OUT_ - 1);
    const int b = idx >> 18;

    const float sx = (x + 0.5f) * 0.25f - 0.5f;
    const float sy = (y + 0.5f) * 0.25f - 0.5f;
    const int   x0 = (int)floorf(sx);
    const int   y0 = (int)floorf(sy);
    const float fx = sx - (float)x0;
    const float fy = sy - (float)y0;
    const int x0c = min(WM_ - 1, max(0, x0));
    const int x1c = min(WM_ - 1, max(0, x0 + 1));
    const int y0c = min(HM_ - 1, max(0, y0));
    const int y1c = min(HM_ - 1, max(0, y0 + 1));

    const float w00 = (1.0f - fy) * (1.0f - fx);
    const float w01 = (1.0f - fy) * fx;
    const float w10 = fy * (1.0f - fx);
    const float w11 = fy * fx;

    const int o00 = y0c * WM_ + x0c;
    const int o01 = y0c * WM_ + x1c;
    const int o10 = y1c * WM_ + x0c;
    const int o11 = y1c * WM_ + x1c;

    const float* base = Mbuf + (size_t)b * C_ * HW_;
    float best = -3.4e38f;
    int   bi   = 0;
    #pragma unroll 4
    for (int c = 0; c < C_; ++c) {
        const float* mp = base + (size_t)c * HW_;
        float v = w00 * mp[o00] + w01 * mp[o01] + w10 * mp[o10] + w11 * mp[o11];
        if (v > best) { best = v; bi = c; }      // strict > : first max wins
    }
    out[idx] = bi;
}

// ---------------------------------------------------------------------------
extern "C" void kernel_launch(void* const* d_in, const int* in_sizes, int n_in,
                              void* d_out, int out_size, void* d_ws, size_t ws_size,
                              hipStream_t stream) {
    const float* logits = (const float*)d_in[0];   // [4,100,80]
    // d_in[1] pred_boxes unused (as in reference)
    const float* masks  = (const float*)d_in[2];   // [4,100,128,128]
    // d_in[3..6] sizes are all 512 (compile-time constants here)

    float*    Mbuf  = (float*)d_ws;
    _Float16* afrag = (_Float16*)((char*)d_ws + M_BYTES);
    int*      out   = (int*)d_out;

    // K1: assignment matrix (tiny)
    segpost_build_assign<<<B_, 128, 0, stream>>>(logits, afrag);

    // K2: WMMA segment-mean. 4096 waves (= B * 1024 N-tiles), 8 waves/block.
    segpost_wmma_segmean<<<(B_ * 1024) / 8, 256, 0, stream>>>(masks, afrag, Mbuf);

    // K3: fused upsample + argmax. B*512*512 threads.
    segpost_upsample_argmax<<<(B_ * OUT_ * OUT_) / 256, 256, 0, stream>>>(Mbuf, out);
}
